// LSTM_5626407157824
// MI455X (gfx1250) — matrix operand order repaired and checked by reading, not compiled, not run on hardware
//
#include <hip/hip_runtime.h>
#include <hip/hip_bf16.h>

typedef __attribute__((ext_vector_type(16))) _Float16 v16h;
typedef __attribute__((ext_vector_type(8)))  _Float16 v8h;
typedef __attribute__((ext_vector_type(8)))  float    v8f;

#define B_TOT 4096
#define LSEQ  336
#define NIN   7
#define NH    16
#define NEXOG 6
#define HOR   24

// gfx1250 has a native v_tanh_f32 trans op (confirmed in disasm). Route BOTH
// activations through it: tanh is 1 op, sigmoid(x) = 0.5*tanh(0.5x)+0.5 is
// mul + tanh + fma (3 ops, 1 trans) instead of mul+exp+add+rcp (4 ops, 2 trans).
#if __has_builtin(__builtin_amdgcn_tanhf)
#define HAVE_NATIVE_TANH 1
#elif __has_builtin(__builtin_amdgcn_tanh_f32)
#define HAVE_NATIVE_TANH 2
#endif

__device__ __forceinline__ float ftanh(float x) {
#if HAVE_NATIVE_TANH == 1
    return __builtin_amdgcn_tanhf(x);
#elif HAVE_NATIVE_TANH == 2
    return __builtin_amdgcn_tanh_f32(x);
#else
    return 2.0f * __builtin_amdgcn_rcpf(1.0f + __expf(-2.0f * x)) - 1.0f;
#endif
}
__device__ __forceinline__ float fsigmoid(float x) {
#ifdef HAVE_NATIVE_TANH
    return __builtin_fmaf(0.5f, ftanh(0.5f * x), 0.5f);
#else
    return __builtin_amdgcn_rcpf(1.0f + __expf(-x));
#endif
}

// One wave (32 lanes) per 16-row batch tile. WMMA 16x16x32 f16:
//   A: M=batch(16) x K=32 = [input(16 padded) | hidden(16)]
//   B: K=32 x N=16, 4 tiles over the 64 gates (i,f,g,o); bias pre-loaded in C
//   D: M=batch x N=gate-slice, f32
// A lane mapping (ISA 7.12.2): lane L holds row M=L%16; half j -> K = j + 8*(L/16) + (j>=8 ? 8 : 0)
// B lane mapping: lane L holds col N=L%16; half j -> K = j + 8*(L/16) + (j>=8 ? 8 : 0), same as A
// D lane mapping: lane L, vgpr v -> M = v + 8*(L/16), N = L%16
__global__ __launch_bounds__(32) void lstm_forecast_kernel(
    const float* __restrict__ x,
    const float* __restrict__ w_ih_l0, const float* __restrict__ w_hh_l0,
    const float* __restrict__ b_ih_l0, const float* __restrict__ b_hh_l0,
    const float* __restrict__ w_ih_l1, const float* __restrict__ w_hh_l1,
    const float* __restrict__ b_ih_l1, const float* __restrict__ b_hh_l1,
    const float* __restrict__ w_fc,   const float* __restrict__ b_fc,
    float* __restrict__ out)
{
    __shared__ __align__(16) _Float16 h0_lds[NH * NH];      // h state layer0, [m][k] row-major
    __shared__ __align__(16) _Float16 h1_lds[NH * NH];      // h state layer1
    __shared__ __align__(16) _Float16 pred_lds[NH * HOR];   // autoregressive feedback

    const int lane   = threadIdx.x;     // 0..31
    const int n      = lane & 15;
    const int g      = lane >> 4;
    const int batch0 = blockIdx.x * 16;

    // ---- preload weights into WMMA B layout (f32 -> f16) ----
    v16h B0[4], B1[4];
    v8f  Cb0[4], Cb1[4];                // bias splat, used as WMMA C operand
#pragma unroll
    for (int t = 0; t < 4; ++t) {
        const int gate = t * 16 + n;
#pragma unroll
        for (int j = 0; j < 16; ++j) {
            const int K = (j < 8) ? (j + 8 * g) : (16 + 8 * g + (j - 8));
            float w0 = (K < NIN) ? w_ih_l0[gate * NIN + K]
                     : (K >= 16) ? w_hh_l0[gate * NH + (K - 16)] : 0.0f;
            float w1 = (K < 16)  ? w_ih_l1[gate * NH + K]
                                 : w_hh_l1[gate * NH + (K - 16)];
            B0[t][j] = (_Float16)w0;
            B1[t][j] = (_Float16)w1;
        }
        const float bs0 = b_ih_l0[gate] + b_hh_l0[gate];
        const float bs1 = b_ih_l1[gate] + b_hh_l1[gate];
#pragma unroll
        for (int j = 0; j < 8; ++j) { Cb0[t][j] = bs0; Cb1[t][j] = bs1; }
    }
    float wfc[NH];
#pragma unroll
    for (int k = 0; k < NH; ++k) wfc[k] = w_fc[k];
    const float bfc = b_fc[0];

    // exog features of the last real row (used by all synthesized rows)
    _Float16 ex[NEXOG];
    {
        const float* lastrow = x + ((size_t)(batch0 + n) * LSEQ + (LSEQ - 1)) * NIN;
#pragma unroll
        for (int e = 0; e < NEXOG; ++e) ex[e] = (_Float16)lastrow[e];
    }

#pragma unroll 1
    for (int s = 0; s < HOR; ++s) {
        // zero h/c state for this horizon step
        v8h hz;
#pragma unroll
        for (int j = 0; j < 8; ++j) hz[j] = (_Float16)0.0f;
        ((v8h*)h0_lds)[lane] = hz;
        ((v8h*)h1_lds)[lane] = hz;
        float c0[8], c1[8];
#pragma unroll
        for (int v = 0; v < 8; ++v) { c0[v] = 0.0f; c1[v] = 0.0f; }
        __syncthreads();

#pragma unroll 1
        for (int t = 0; t < LSEQ; ++t) {
            const int r = s + t;

            // ======== layer 0 ========
            v16h a0;
            if (g == 0) {
                if (r < LSEQ) {
                    const float* xr = x + ((size_t)(batch0 + n) * LSEQ + r) * NIN;
#pragma unroll
                    for (int j = 0; j < NIN; ++j) a0[j] = (_Float16)xr[j];
                    a0[7] = (_Float16)0.0f;
                } else {
#pragma unroll
                    for (int e = 0; e < NEXOG; ++e) a0[e] = ex[e];
                    a0[6] = pred_lds[n * HOR + (r - LSEQ)];
                    a0[7] = (_Float16)0.0f;
                }
            } else {
#pragma unroll
                for (int j = 0; j < 8; ++j) a0[j] = (_Float16)0.0f;
            }
            {
                v8h hp = *(const v8h*)&h0_lds[n * NH + g * 8];
#pragma unroll
                for (int j = 0; j < 8; ++j) a0[8 + j] = hp[j];
            }

            v8f acc0[4];
#pragma unroll
            for (int q = 0; q < 4; ++q) {
                acc0[q] = __builtin_amdgcn_wmma_f32_16x16x32_f16(
                    false, a0, false, B0[q], (short)0, Cb0[q], false, false);
            }

            _Float16 h0new[8];
#pragma unroll
            for (int v = 0; v < 8; ++v) {
                float iv = fsigmoid(acc0[0][v]);
                float fv = fsigmoid(acc0[1][v]);
                float gv = ftanh   (acc0[2][v]);
                float ov = fsigmoid(acc0[3][v]);
                float c  = fv * c0[v] + iv * gv;
                c0[v] = c;
                h0new[v] = (_Float16)(ov * ftanh(c));
            }
            __syncthreads();
#pragma unroll
            for (int v = 0; v < 8; ++v)
                h0_lds[(v + 8 * g) * NH + n] = h0new[v];
            __syncthreads();

            // ======== layer 1 ========
            v16h a1;
            {
                v8h inlo = *(const v8h*)&h0_lds[n * NH + g * 8];
                v8h sthi = *(const v8h*)&h1_lds[n * NH + g * 8];
#pragma unroll
                for (int j = 0; j < 8; ++j) { a1[j] = inlo[j]; a1[8 + j] = sthi[j]; }
            }

            v8f acc1[4];
#pragma unroll
            for (int q = 0; q < 4; ++q) {
                acc1[q] = __builtin_amdgcn_wmma_f32_16x16x32_f16(
                    false, a1, false, B1[q], (short)0, Cb1[q], false, false);
            }

            _Float16 h1new[8];
#pragma unroll
            for (int v = 0; v < 8; ++v) {
                float iv = fsigmoid(acc1[0][v]);
                float fv = fsigmoid(acc1[1][v]);
                float gv = ftanh   (acc1[2][v]);
                float ov = fsigmoid(acc1[3][v]);
                float c  = fv * c1[v] + iv * gv;
                c1[v] = c;
                h1new[v] = (_Float16)(ov * ftanh(c));
            }
            __syncthreads();
#pragma unroll
            for (int v = 0; v < 8; ++v)
                h1_lds[(v + 8 * g) * NH + n] = h1new[v];
            __syncthreads();
        }

        // ======== FC head + autoregressive feedback ========
        if (lane < 16) {
            float sum = bfc;
#pragma unroll
            for (int k = 0; k < NH; ++k) sum += (float)h1_lds[n * NH + k] * wfc[k];
            out[(size_t)(batch0 + n) * HOR + s] = sum;
            pred_lds[n * HOR + s] = (_Float16)sum;
        }
        __syncthreads();
    }
}

extern "C" void kernel_launch(void* const* d_in, const int* in_sizes, int n_in,
                              void* d_out, int out_size, void* d_ws, size_t ws_size,
                              hipStream_t stream) {
    (void)in_sizes; (void)n_in; (void)d_ws; (void)ws_size; (void)out_size;
    const float* x        = (const float*)d_in[0];
    const float* w_ih_l0  = (const float*)d_in[1];
    const float* w_hh_l0  = (const float*)d_in[2];
    const float* b_ih_l0  = (const float*)d_in[3];
    const float* b_hh_l0  = (const float*)d_in[4];
    const float* w_ih_l1  = (const float*)d_in[5];
    const float* w_hh_l1  = (const float*)d_in[6];
    const float* b_ih_l1  = (const float*)d_in[7];
    const float* b_hh_l1  = (const float*)d_in[8];
    const float* w_fc     = (const float*)d_in[9];
    const float* b_fc     = (const float*)d_in[10];
    float* out            = (float*)d_out;

    dim3 grid(B_TOT / 16);   // 256 independent batch tiles, one wave each
    dim3 block(32);
    lstm_forecast_kernel<<<grid, block, 0, stream>>>(
        x, w_ih_l0, w_hh_l0, b_ih_l0, b_hh_l0,
        w_ih_l1, w_hh_l1, b_ih_l1, b_hh_l1,
        w_fc, b_fc, out);
}
